// TabNetRegressor_80298708566461
// MI455X (gfx1250) — matrix operand order repaired and checked by reading, not compiled, not run on hardware
//
#include <hip/hip_runtime.h>
#include <hip/hip_bf16.h>
#include <math.h>

typedef __attribute__((ext_vector_type(16))) __bf16 v16bf;
typedef __attribute__((ext_vector_type(8)))  float  v8f;

#define BN_EPS 1e-5f
#define RES_SCALE 0.70710678118654752f
#define GAMMA 1.5f

// Wave-local LDS fence: each wave owns a private LDS slice, so instead of a
// workgroup s_barrier (which couples 4 independent waves), we only need
// in-wave LDS ordering: DScnt drain + compiler memory barrier.
#define WAVE_SYNC() asm volatile("s_wait_dscnt 0x0" ::: "memory")

// ---------------------------------------------------------------------------
// Weight prep: f32 (K x N, row-major, nmat batched) -> bf16 (N x K per matrix)
// ---------------------------------------------------------------------------
__global__ void convt_kernel(const float* __restrict__ src, __bf16* __restrict__ dst,
                             int K, int N, int nmat) {
    int i = blockIdx.x * blockDim.x + threadIdx.x;
    int total = nmat * K * N;
    if (i >= total) return;
    int mat = i / (K * N);
    int rem = i - mat * K * N;
    int k = rem / N;
    int n = rem - k * N;
    dst[(size_t)mat * N * K + (size_t)n * K + k] = (__bf16)src[i];
}

// ---------------------------------------------------------------------------
// WMMA helpers
// ---------------------------------------------------------------------------
static __device__ __forceinline__ v8f wmma_bf16(v16bf a, v16bf b, v8f c) {
    return __builtin_amdgcn_wmma_f32_16x16x32_bf16(false, a, false, b, (short)0, c,
                                                   false, false);
}

// A fragment: 16x32 bf16 tile from row-major [16 x K] buffer (LDS).
// lane m = lane&15 holds row m; hk = lane>>4 selects K-half:
//   elems 0..15 = K = kbase + hk*16 + 0..15
static __device__ __forceinline__ v16bf load_a_frag(const __bf16* __restrict__ src,
                                                    int K, int kbase, int lane) {
    int m = lane & 15, hk = lane >> 4;
    union { v16bf v; uint4 q[2]; } u;
    u.q[0] = *(const uint4*)(src + (size_t)m * K + kbase + hk * 16);
    u.q[1] = *(const uint4*)(src + (size_t)m * K + kbase + hk * 16 + 8);
    return u.v;
}

// B fragment: 32x16 bf16 tile from N-major bf16 weights WT[n][k] (global).
// lane n = lane&15 holds column ncol_base+n; hk = lane>>4:
//   elems 0..15 = K = kbase + hk*16 + 0..15 (contiguous 32 bytes)
static __device__ __forceinline__ v16bf load_b_frag(const __bf16* __restrict__ wt,
                                                    int K, int ncol_base, int kbase,
                                                    int lane) {
    int n = lane & 15, hk = lane >> 4;
    const __bf16* p = wt + (size_t)(ncol_base + n) * K + kbase + hk * 16;
    union { v16bf v; uint4 q[2]; } u;
    u.q[0] = ((const uint4*)p)[0];
    u.q[1] = ((const uint4*)p)[1];
    return u.v;
}

// ---------------------------------------------------------------------------
// One GLU layer: dst(16x128) = glu( bn( src(16xK) @ W(Kx256) + b ) )
// residual: dst = (src + glu)*sqrt(0.5)  (requires K==128)
// C-fragment layout: lane -> col n=lane&15, element r -> row 8*(lane>>4)+r
// ---------------------------------------------------------------------------
static __device__ __forceinline__ void glu_layer(
    const __bf16* __restrict__ src, int K,
    const __bf16* __restrict__ WT,      // 256 x K, N-major, bf16
    const float* __restrict__ bias,     // 256
    const float* __restrict__ bn,       // 4 x 256 (s,b,m,v)
    __bf16* __restrict__ dst,           // 16 x 128
    bool residual, int lane) {
    int n = lane & 15, h = lane >> 4;
    int nk = K >> 5;
    for (int t = 0; t < 8; ++t) {
        v8f co = {}; v8f cg = {};
        for (int kc = 0; kc < nk; ++kc) {
            v16bf a  = load_a_frag(src, K, kc * 32, lane);
            v16bf b0 = load_b_frag(WT, K, t * 16, kc * 32, lane);
            co = wmma_bf16(a, b0, co);
            v16bf b1 = load_b_frag(WT, K, 128 + t * 16, kc * 32, lane);
            cg = wmma_bf16(a, b1, cg);
        }
        int oc = t * 16 + n;       // o column (0..127)
        int gc = 128 + oc;         // g column (128..255)
        float osc = rsqrtf(bn[768 + oc] + BN_EPS) * bn[oc];
        float gsc = rsqrtf(bn[768 + gc] + BN_EPS) * bn[gc];
        float oof = bias[oc] - bn[512 + oc];
        float gof = bias[gc] - bn[512 + gc];
        float obb = bn[256 + oc], gbb = bn[256 + gc];
        #pragma unroll
        for (int r = 0; r < 8; ++r) {
            float ov = (co[r] + oof) * osc + obb;
            float gv = (cg[r] + gof) * gsc + gbb;
            float y = ov * (1.0f / (1.0f + __expf(-gv)));
            int mrow = h * 8 + r;
            if (residual) {
                float hin = (float)src[(size_t)mrow * K + oc];
                y = (hin + y) * RES_SCALE;
            }
            dst[(size_t)mrow * 128 + oc] = (__bf16)y;
        }
    }
}

// ---------------------------------------------------------------------------
// Main TabNet kernel: one wave per 16-row tile, 4 waves per block.
// Per-wave LDS slice (bytes): x 16384 | xm 16384 | h0 4096 | h1 4096 |
//                             abuf 2048 | logit 5120 | prior 5120 | dsum 4096
// Waves are fully independent: only wave-local LDS fences, no s_barrier.
// ---------------------------------------------------------------------------
#define WAVE_LDS 57344
#define WAVES_PER_BLOCK 4

__global__ __launch_bounds__(128, 1) void tabnet_kernel(
    const float* __restrict__ x_num, const int* __restrict__ x_cat,
    const float* __restrict__ emb_W, const float* __restrict__ in_bn,
    const float* __restrict__ sh_b0, const float* __restrict__ sh_bn0,
    const float* __restrict__ sh_b1, const float* __restrict__ sh_bn1,
    const float* __restrict__ init_b, const float* __restrict__ init_bn,
    const float* __restrict__ step_b, const float* __restrict__ step_bn,
    const float* __restrict__ att_b, const float* __restrict__ att_bn,
    const float* __restrict__ out_W, const float* __restrict__ out_b,
    const __bf16* __restrict__ WT0,     // 256 x 512
    const __bf16* __restrict__ WT1,     // 256 x 128
    const __bf16* __restrict__ initWT,  // 2 x 256 x 128
    const __bf16* __restrict__ stepWT,  // 10 x 256 x 128
    const __bf16* __restrict__ attWT,   // 5 x 80 x 64
    float* __restrict__ out) {
    extern __shared__ char smem[];
    int lane = threadIdx.x & 31;
    int wave = threadIdx.x >> 5;
    char* base = smem + (size_t)wave * WAVE_LDS;
    __bf16* xw    = (__bf16*)(base);                 // 16 x 512
    __bf16* xm    = (__bf16*)(base + 16384);         // 16 x 512
    __bf16* h0    = (__bf16*)(base + 32768);         // 16 x 128
    __bf16* h1    = (__bf16*)(base + 36864);         // 16 x 128
    __bf16* abuf  = (__bf16*)(base + 40960);         // 16 x 64
    float*  logit = (float*) (base + 43008);         // 16 x 80 (mask in-place)
    float*  prior = (float*) (base + 48128);         // 16 x 80
    float*  dsum  = (float*) (base + 53248);         // 16 x 64

    int tile = blockIdx.x * WAVES_PER_BLOCK + wave;
    int row0 = tile * 16;

    // ---- build x = bn(concat(x_num, emb)) ----
    for (int e = lane; e < 16 * 512; e += 32) {
        int m = e >> 9, f = e & 511;
        int row = row0 + m;
        float val;
        if (f < 64) {
            val = x_num[(size_t)row * 64 + f];
        } else {
            int ci = (f - 64) >> 4, ei = (f - 64) & 15;
            int idx = x_cat[(size_t)row * 16 + ci];
            val = emb_W[((size_t)ci * 1000 + idx) * 16 + ei];
        }
        val = (val - in_bn[1024 + f]) * rsqrtf(in_bn[1536 + f] + BN_EPS) * in_bn[f]
              + in_bn[512 + f];
        xw[e] = (__bf16)val;
    }
    for (int e = lane; e < 16 * 80; e += 32) prior[e] = 1.0f;
    for (int e = lane; e < 16 * 64; e += 32) dsum[e] = 0.0f;
    WAVE_SYNC();

    // ---- initial pass: shared + init indep ----
    glu_layer(xw, 512, WT0, sh_b0, sh_bn0, h0, false, lane); WAVE_SYNC();
    glu_layer(h0, 128, WT1, sh_b1, sh_bn1, h1, true, lane);  WAVE_SYNC();
    glu_layer(h1, 128, initWT,             init_b,       init_bn,        h0, true, lane); WAVE_SYNC();
    glu_layer(h0, 128, initWT + 256 * 128, init_b + 256, init_bn + 1024, h1, true, lane); WAVE_SYNC();
    for (int e = lane; e < 16 * 64; e += 32) {
        int m = e >> 6, c = e & 63;
        abuf[e] = h1[(size_t)m * 128 + 64 + c];
    }
    WAVE_SYNC();

    // ---- 5 decision steps ----
    for (int step = 0; step < 5; ++step) {
        // attention: logits = bn(a @ att_W + att_b) * prior   (16x80, 5 N-tiles)
        {
            int n = lane & 15, h = lane >> 4;
            const __bf16* wt = attWT + (size_t)step * 80 * 64;
            const float* bnp = att_bn + (size_t)step * 320;
            for (int t = 0; t < 5; ++t) {
                v8f c = {};
                for (int kc = 0; kc < 2; ++kc) {
                    v16bf a = load_a_frag(abuf, 64, kc * 32, lane);
                    v16bf b = load_b_frag(wt, 64, t * 16, kc * 32, lane);
                    c = wmma_bf16(a, b, c);
                }
                int col = t * 16 + n;
                float sc = rsqrtf(bnp[240 + col] + BN_EPS) * bnp[col];
                float of = att_b[step * 80 + col] - bnp[160 + col];
                float bb = bnp[80 + col];
                #pragma unroll
                for (int r = 0; r < 8; ++r) {
                    int mrow = h * 8 + r;
                    float v = ((c[r] + of) * sc + bb) * prior[mrow * 80 + col];
                    logit[mrow * 80 + col] = v;
                }
            }
        }
        WAVE_SYNC();

        // sparsemax per row via bisection on tau; update prior (lanes 0..15)
        if (lane < 16) {
            float* z = logit + lane * 80;
            float zmax = -1e30f;
            for (int j = 0; j < 80; ++j) zmax = fmaxf(zmax, z[j]);
            float lo = zmax - 1.0f, hi = zmax;
            for (int it = 0; it < 30; ++it) {
                float mid = 0.5f * (lo + hi), s = 0.0f;
                for (int j = 0; j < 80; ++j) s += fmaxf(z[j] - mid, 0.0f);
                if (s > 1.0f) lo = mid; else hi = mid;
            }
            float tau = 0.5f * (lo + hi);
            float* pr = prior + lane * 80;
            for (int j = 0; j < 80; ++j) {
                float mv = fmaxf(z[j] - tau, 0.0f);
                z[j] = mv;                      // mask_group in-place
                pr[j] *= (GAMMA - mv);
            }
        }
        WAVE_SYNC();

        // masked input: xm = x * mask_group[GROUP_IDX(f)]
        for (int e = lane; e < 16 * 512; e += 32) {
            int m = e >> 9, f = e & 511;
            int g = (f < 64) ? f : 64 + ((f - 64) >> 4);
            xm[e] = (__bf16)((float)xw[e] * logit[m * 80 + g]);
        }
        WAVE_SYNC();

        // feature transformer: shared + step-specific indep
        glu_layer(xm, 512, WT0, sh_b0, sh_bn0, h0, false, lane); WAVE_SYNC();
        glu_layer(h0, 128, WT1, sh_b1, sh_bn1, h1, true, lane);  WAVE_SYNC();
        int l0 = step * 2, l1 = step * 2 + 1;
        glu_layer(h1, 128, stepWT + (size_t)l0 * 32768, step_b + l0 * 256,
                  step_bn + l0 * 1024, h0, true, lane); WAVE_SYNC();
        glu_layer(h0, 128, stepWT + (size_t)l1 * 32768, step_b + l1 * 256,
                  step_bn + l1 * 1024, h1, true, lane); WAVE_SYNC();

        // decision_sum += relu(d); a = out[:,64:128]
        for (int e = lane; e < 16 * 64; e += 32) {
            int m = e >> 6, c = e & 63;
            dsum[e] += fmaxf((float)h1[(size_t)m * 128 + c], 0.0f);
            abuf[e] = h1[(size_t)m * 128 + 64 + c];
        }
        WAVE_SYNC();
    }

    // ---- final: out = decision_sum @ out_W + out_b ----
    if (lane < 16) {
        float s = out_b[0];
        const float* d = dsum + lane * 64;
        for (int c = 0; c < 64; ++c) s += d[c] * out_W[c];
        out[row0 + lane] = s;
    }
}

// ---------------------------------------------------------------------------
extern "C" void kernel_launch(void* const* d_in, const int* in_sizes, int n_in,
                              void* d_out, int out_size, void* d_ws, size_t ws_size,
                              hipStream_t stream) {
    const float* x_num  = (const float*)d_in[0];
    const int*   x_cat  = (const int*)  d_in[1];
    const float* emb_W  = (const float*)d_in[2];
    const float* in_bn  = (const float*)d_in[3];
    const float* sh_W0  = (const float*)d_in[4];
    const float* sh_b0  = (const float*)d_in[5];
    const float* sh_bn0 = (const float*)d_in[6];
    const float* sh_W1  = (const float*)d_in[7];
    const float* sh_b1  = (const float*)d_in[8];
    const float* sh_bn1 = (const float*)d_in[9];
    const float* init_W = (const float*)d_in[10];
    const float* init_b = (const float*)d_in[11];
    const float* init_bn= (const float*)d_in[12];
    const float* step_W = (const float*)d_in[13];
    const float* step_b = (const float*)d_in[14];
    const float* step_bn= (const float*)d_in[15];
    const float* att_W  = (const float*)d_in[16];
    const float* att_b  = (const float*)d_in[17];
    const float* att_bn = (const float*)d_in[18];
    const float* out_W  = (const float*)d_in[19];
    const float* out_b  = (const float*)d_in[20];

    // workspace: transposed bf16 weights
    char* w = (char*)d_ws;
    __bf16* WT0    = (__bf16*)(w);                    // 256x512
    __bf16* WT1    = (__bf16*)(w + 262144);           // 256x128
    __bf16* initWT = (__bf16*)(w + 327680);           // 2 x 256x128
    __bf16* stepWT = (__bf16*)(w + 458752);           // 10 x 256x128
    __bf16* attWT  = (__bf16*)(w + 1114112);          // 5 x 80x64

    const int CT = 256;
    convt_kernel<<<(512 * 256 + CT - 1) / CT, CT, 0, stream>>>(sh_W0, WT0, 512, 256, 1);
    convt_kernel<<<(128 * 256 + CT - 1) / CT, CT, 0, stream>>>(sh_W1, WT1, 128, 256, 1);
    convt_kernel<<<(2 * 128 * 256 + CT - 1) / CT, CT, 0, stream>>>(init_W, initWT, 128, 256, 2);
    convt_kernel<<<(10 * 128 * 256 + CT - 1) / CT, CT, 0, stream>>>(step_W, stepWT, 128, 256, 10);
    convt_kernel<<<(5 * 64 * 80 + CT - 1) / CT, CT, 0, stream>>>(att_W, attWT, 64, 80, 5);

    size_t lds_bytes = (size_t)WAVE_LDS * WAVES_PER_BLOCK;   // 229376 B
    (void)hipFuncSetAttribute((const void*)tabnet_kernel,
                              hipFuncAttributeMaxDynamicSharedMemorySize,
                              (int)lds_bytes);

    int nblocks = 65536 / (16 * WAVES_PER_BLOCK);            // 1024
    tabnet_kernel<<<nblocks, 32 * WAVES_PER_BLOCK, lds_bytes, stream>>>(
        x_num, x_cat, emb_W, in_bn, sh_b0, sh_bn0, sh_b1, sh_bn1,
        init_b, init_bn, step_b, step_bn, att_b, att_bn, out_W, out_b,
        WT0, WT1, initWT, stepWT, attWT, (float*)d_out);
}
